// Update_40570261078233
// MI455X (gfx1250) — matrix operand order repaired and checked by reading, not compiled, not run on hardware
//
#include <hip/hip_runtime.h>

#define NTOT   262144
#define BLKROW 64            // rows per block (4 waves x 16 rows)
#define LDS_PER_WAVE_H 12288 // halves: 12KB regionA (mst+hst, later abc) + 12KB vbuf
#define LDS_BYTES (4 * LDS_PER_WAVE_H * 2)  // 98304 B -> up to 3 WGs (12 waves) per WGP

typedef __attribute__((ext_vector_type(16))) _Float16 v16h;
typedef __attribute__((ext_vector_type(8)))  _Float16 v8h;
typedef __attribute__((ext_vector_type(4)))  _Float16 v4h;
typedef __attribute__((ext_vector_type(8)))  float    v8f;
typedef __attribute__((ext_vector_type(4)))  float    v4f;

static __device__ __forceinline__ v16h hcat(v8h lo, v8h hi) {
  return __builtin_shufflevector(lo, hi, 0,1,2,3,4,5,6,7,8,9,10,11,12,13,14,15);
}

static __device__ __forceinline__ v8f wmma16(v16h a, v16h b, v8f c) {
  // v_wmma_f32_16x16x32_f16: D = A(16x32 f16) * B(32x16 f16) + C(16x16 f32)
  return __builtin_amdgcn_wmma_f32_16x16x32_f16(false, a, false, b, (short)0, c, false, false);
}

// A operand (16x32, f16): lane holds K = 16h + {0..15}, contiguous in memory
static __device__ __forceinline__ v16h loadA(const _Float16* p) {
  v8h lo = *(const v8h*)(p);
  v8h hi = *(const v8h*)(p + 8);
  return hcat(lo, hi);
}
// B operand (32x16, f16): lane (column) holds K = 16h + {0..15}, contiguous in memory
static __device__ __forceinline__ v16h loadB(const _Float16* p) {
  v8h lo = *(const v8h*)(p);
  v8h hi = *(const v8h*)(p + 8);
  return hcat(lo, hi);
}

// ---------------- weight prep: fp32 -> f16 once, into d_ws -------------------
// layout in halves: Uh @ 0 (16384) | Vh @ 16384 (16384) | W1h @ 32768 (32768) | W2h @ 65536 (49152)
__global__ void prep_weights(const float* __restrict__ U, const float* __restrict__ V,
                             const float* __restrict__ W1, const float* __restrict__ W2,
                             _Float16* __restrict__ out) {
  int i = blockIdx.x * 256 + threadIdx.x;
  if (i < 16384)       out[i] = (_Float16)U[i];
  else if (i < 32768)  out[i] = (_Float16)V[i - 16384];
  else if (i < 65536)  out[i] = (_Float16)W1[i - 32768];
  else if (i < 114688) out[i] = (_Float16)W2[i - 65536];
}

// ---------------- fused update kernel ---------------------------------------
__global__ __launch_bounds__(128)
void fused_update(const float* __restrict__ vec, const float* __restrict__ scal,
                  const float* __restrict__ b1, const float* __restrict__ b2,
                  const _Float16* __restrict__ wh, float* __restrict__ out) {
  const int tid  = threadIdx.x;
  const int wave = tid >> 5;
  const int lane = tid & 31;
  const int hh   = lane >> 4;   // half-wave select
  const int M    = lane & 15;   // row (A) / column (B,C,D) within tile
  const int n0   = blockIdx.x * BLKROW + wave * 16;

  extern __shared__ _Float16 smem[];
  _Float16* ws0  = smem + wave * LDS_PER_WAVE_H; // per-wave private slice, no barriers
  // region A (ws0[0..6143]) time-multiplexed:
  //   t0: vecstage f16 [3][16][128]            (consumed into Avec registers)
  //   t1: mst [16][256] (m = scalar|vnorm) + hst [16][128]
  //   t2: abc D-layout blob [24 tiles][32 lanes][8]   (after hst -> Ah preload)
  _Float16* mst  = ws0;
  _Float16* hst  = ws0 + 4096;
  _Float16* abc  = ws0;
  _Float16* vbuf = ws0 + 6144;    // v D-layout blob [3*8 tiles][32 lanes][8], live t1..t4

  // ---- stage vec tile: de-interleave [16][128][3] f32 -> f16 [3][16][128] in ws0[0..6143]
#pragma unroll 4
  for (int r = 0; r < 16; ++r) {
    const float* vp = vec + (size_t)(n0 + r) * 384 + lane * 12;  // 4 xyz-triples per lane
    v4f f0 = *(const v4f*)(vp);
    v4f f1 = *(const v4f*)(vp + 4);
    v4f f2 = *(const v4f*)(vp + 8);
    v4h a0 = {(_Float16)f0[0], (_Float16)f0[3], (_Float16)f1[2], (_Float16)f2[1]};
    v4h a1 = {(_Float16)f0[1], (_Float16)f1[0], (_Float16)f1[3], (_Float16)f2[2]};
    v4h a2 = {(_Float16)f0[2], (_Float16)f1[1], (_Float16)f2[0], (_Float16)f2[3]};
    _Float16* dst = ws0 + r * 128 + lane * 4;
    *(v4h*)(dst)        = a0;
    *(v4h*)(dst + 2048) = a1;
    *(v4h*)(dst + 4096) = a2;
  }

  // ---- gather the 12 A-operands (3 components x 4 K-chunks); kept in VGPRs for V and U GEMMs
  v16h Avec[3][4];
#pragma unroll
  for (int i = 0; i < 3; ++i)
#pragma unroll
    for (int kc = 0; kc < 4; ++kc)
      Avec[i][kc] = loadA(ws0 + i * 2048 + M * 128 + kc * 32 + 16 * hh);

  // region A now reusable: stage scalar_embeddings into m[.,0:128] (f16)
#pragma unroll 4
  for (int r = 0; r < 16; ++r) {
    v4f s = *(const v4f*)(scal + (size_t)(n0 + r) * 128 + lane * 4);
    v4h p = {(_Float16)s[0], (_Float16)s[1], (_Float16)s[2], (_Float16)s[3]};
    *(v4h*)(mst + r * 256 + lane * 4) = p;
  }

  // ---- phase 1: v = V_w @ x  (3 GEMMs), vnorm -> m[.,128:256], v -> vbuf (f16 D-blob)
  const _Float16* Vh = wh + 16384;
#pragma unroll 1
  for (int ot = 0; ot < 8; ++ot) {
    const int o = ot * 16 + M;
    v8f v0 = {}, v1 = {}, v2 = {};
#pragma unroll
    for (int kc = 0; kc < 4; ++kc) {
      v16h B = loadB(Vh + o * 128 + kc * 32 + 16 * hh);
      v0 = wmma16(Avec[0][kc], B, v0);
      v1 = wmma16(Avec[1][kc], B, v1);
      v2 = wmma16(Avec[2][kc], B, v2);
    }
    v8h p0, p1, p2;
#pragma unroll
    for (int j = 0; j < 8; ++j) {
      float ss = v0[j]*v0[j] + v1[j]*v1[j] + v2[j]*v2[j] + 1e-8f;
      mst[(j + 8*hh) * 256 + 128 + o] = (_Float16)sqrtf(ss);
      p0[j] = (_Float16)v0[j]; p1[j] = (_Float16)v1[j]; p2[j] = (_Float16)v2[j];
    }
    *(v8h*)(vbuf + ((0*8 + ot) * 32 + lane) * 8) = p0;
    *(v8h*)(vbuf + ((1*8 + ot) * 32 + lane) * 8) = p1;
    *(v8h*)(vbuf + ((2*8 + ot) * 32 + lane) * 8) = p2;
  }

  // ---- phase 2: h = silu(m @ W1^T + b1),  K = 256
  const _Float16* W1h = wh + 32768;
#pragma unroll 1
  for (int ot = 0; ot < 8; ++ot) {
    const int o = ot * 16 + M;
    const float bc = b1[o];
    v8f acc = {};
#pragma unroll
    for (int kc = 0; kc < 8; ++kc) {
      v16h A = loadA(mst + M * 256 + kc * 32 + 16 * hh);
      v16h B = loadB(W1h + o * 256 + kc * 32 + 16 * hh);
      acc = wmma16(A, B, acc);
    }
#pragma unroll
    for (int j = 0; j < 8; ++j) {
      float x = acc[j] + bc;
      float s = 1.0f / (1.0f + __expf(-x));
      hst[(j + 8*hh) * 128 + o] = (_Float16)(x * s);
    }
  }

  // ---- phase 3: [a|b|c] = h @ W2^T + b2,  24 output tiles -> abc blob
  // Preload all A-operands from hst FIRST: abc aliases the mst/hst region.
  const _Float16* W2h = wh + 65536;
  v16h Ah[4];
#pragma unroll
  for (int kc = 0; kc < 4; ++kc) Ah[kc] = loadA(hst + M * 128 + kc * 32 + 16 * hh);
#pragma unroll 1
  for (int ot = 0; ot < 24; ++ot) {
    const int o = ot * 16 + M;
    const float bc = b2[o];
    v8f acc = {};
#pragma unroll
    for (int kc = 0; kc < 4; ++kc)
      acc = wmma16(Ah[kc], loadB(W2h + o * 128 + kc * 32 + 16 * hh), acc);
    v8h p;
#pragma unroll
    for (int j = 0; j < 8; ++j) p[j] = (_Float16)(acc[j] + bc);
    *(v8h*)(abc + (ot * 32 + lane) * 8) = p;
  }

  // ---- phase 4: u = U_w @ x (reuse Avec), epilogue, stores
  float* dvo = out;                          // [N][128][3]
  float* dso = out + (size_t)NTOT * 384;     // [N][128]
  const _Float16* Uh = wh;
#pragma unroll 1
  for (int ot = 0; ot < 8; ++ot) {
    const int o = ot * 16 + M;
    v8f u0 = {}, u1 = {}, u2 = {};
#pragma unroll
    for (int kc = 0; kc < 4; ++kc) {
      v16h B = loadB(Uh + o * 128 + kc * 32 + 16 * hh);
      u0 = wmma16(Avec[0][kc], B, u0);
      u1 = wmma16(Avec[1][kc], B, u1);
      u2 = wmma16(Avec[2][kc], B, u2);
    }
    v8h ahv = *(const v8h*)(abc + (( 0 + ot) * 32 + lane) * 8);
    v8h bhv = *(const v8h*)(abc + (( 8 + ot) * 32 + lane) * 8);
    v8h chv = *(const v8h*)(abc + ((16 + ot) * 32 + lane) * 8);
    v8h w0  = *(const v8h*)(vbuf + ((0*8 + ot) * 32 + lane) * 8);
    v8h w1  = *(const v8h*)(vbuf + ((1*8 + ot) * 32 + lane) * 8);
    v8h w2  = *(const v8h*)(vbuf + ((2*8 + ot) * 32 + lane) * 8);
#pragma unroll
    for (int j = 0; j < 8; ++j) {
      const int n = n0 + j + 8 * hh;
      float av = (float)ahv[j], bv = (float)bhv[j], cv = (float)chv[j];
      float x0 = u0[j], x1 = u1[j], x2 = u2[j];
      size_t di = ((size_t)n * 128 + o) * 3;
      dvo[di + 0] = x0 * av;
      dvo[di + 1] = x1 * av;
      dvo[di + 2] = x2 * av;
      float dot = x0 * (float)w0[j] + x1 * (float)w1[j] + x2 * (float)w2[j];
      dso[(size_t)n * 128 + o] = bv + cv * dot;
    }
  }
}

extern "C" void kernel_launch(void* const* d_in, const int* in_sizes, int n_in,
                              void* d_out, int out_size, void* d_ws, size_t ws_size,
                              hipStream_t stream) {
  (void)in_sizes; (void)n_in; (void)out_size; (void)ws_size;
  const float* vec  = (const float*)d_in[0];
  const float* scal = (const float*)d_in[1];
  const float* U    = (const float*)d_in[2];
  const float* Vw   = (const float*)d_in[3];
  const float* W1   = (const float*)d_in[4];
  const float* b1   = (const float*)d_in[5];
  const float* W2   = (const float*)d_in[6];
  const float* b2   = (const float*)d_in[7];
  _Float16* wh = (_Float16*)d_ws;   // 229376 bytes of f16 weights

  prep_weights<<<448, 256, 0, stream>>>(U, Vw, W1, W2, wh);
  fused_update<<<NTOT / BLKROW, 128, LDS_BYTES, stream>>>(vec, scal, b1, b2, wh, (float*)d_out);
}
